// MoE_54185307406693
// MI455X (gfx1250) — matrix operand order repaired and checked by reading, not compiled, not run on hardware
//
#include <hip/hip_runtime.h>
#include <math.h>

// Problem constants (B,S,D,H,N,K,Ks = 2,1024,1024,1024,8,2,1)
#define M_TOK   2048          // B*S tokens
#define DDIM    1024
#define HDIM    1024
#define TWOH    2048
#define NEXP    8
#define SHARED_E (NEXP - 1)
#define LSTR    40            // LDS row stride (halves): 16B-aligned, bank-skewed

typedef float v8f  __attribute__((ext_vector_type(8)));
typedef __bf16 v16bf __attribute__((ext_vector_type(16)));

union Frag { v16bf v; uint4 q[2]; };

#define WMMA_BF16(A, B, C) \
    __builtin_amdgcn_wmma_f32_16x16x32_bf16(false, (A), false, (B), (short)0, (C), false, false)

__device__ __forceinline__ unsigned short f2bf(float f) {
    union { float f; unsigned u; } v; v.f = f;
    unsigned r = v.u + 0x7FFFu + ((v.u >> 16) & 1u);   // round-to-nearest-even
    return (unsigned short)(r >> 16);
}

__device__ __forceinline__ float fastrcp(float x) { return __builtin_amdgcn_rcpf(x); }

// CDNA5 async copy: global -> LDS, 16B per lane, tracked by ASYNCcnt (no VGPR staging)
__device__ __forceinline__ void asyncB128(const unsigned short* g, const unsigned short* l) {
    unsigned           lds = (unsigned)(size_t)l;           // flat LDS addr low 32b = LDS offset
    unsigned long long ga  = (unsigned long long)(size_t)g;
    asm volatile("global_load_async_to_lds_b128 %0, %1, off" :: "v"(lds), "v"(ga) : "memory");
}
__device__ __forceinline__ void waitAsync0() {
    asm volatile("s_wait_asynccnt 0x0" ::: "memory");
}

// ---------------------------------------------------------------- init: zero y, cnt
__global__ void kInit(float* __restrict__ y, int* __restrict__ cnt) {
    int i = blockIdx.x * 256 + threadIdx.x;
    if (i < M_TOK * DDIM / 4) ((float4*)y)[i] = float4{0.f, 0.f, 0.f, 0.f};
    if (i < NEXP) cnt[i] = 0;
}

// ---------------------------------------------------------------- fp32 -> bf16 converts
__global__ void kConvertX(const float* __restrict__ x, unsigned short* __restrict__ xb) {
    int i = blockIdx.x * 256 + threadIdx.x;                 // 8 elements per thread
    if (i >= M_TOK * DDIM / 8) return;
    const float4* p = (const float4*)(x + (size_t)i * 8);
    float4 a = p[0], b = p[1];
    union { unsigned short h[8]; uint4 q; } o;
    o.h[0] = f2bf(a.x); o.h[1] = f2bf(a.y); o.h[2] = f2bf(a.z); o.h[3] = f2bf(a.w);
    o.h[4] = f2bf(b.x); o.h[5] = f2bf(b.y); o.h[6] = f2bf(b.z); o.h[7] = f2bf(b.w);
    *(uint4*)(xb + (size_t)i * 8) = o.q;
}

// Wl1 [e][k][col] -> w1t [e][col][k]  bf16
__global__ void kTransW1(const float* __restrict__ w1, unsigned short* __restrict__ w1t) {
    int i = blockIdx.x * 256 + threadIdx.x;
    if (i >= NEXP * DDIM * TWOH) return;
    int col = i % TWOH;
    int rem = i / TWOH;
    int k   = rem % DDIM;
    int e   = rem / DDIM;
    w1t[((size_t)e * TWOH + col) * DDIM + k] = f2bf(w1[i]);
}

// Wl2 [e][h][d] -> w2t [e][d][h]  bf16
__global__ void kTransW2(const float* __restrict__ w2, unsigned short* __restrict__ w2t) {
    int i = blockIdx.x * 256 + threadIdx.x;
    if (i >= NEXP * HDIM * DDIM) return;
    int d   = i % DDIM;
    int rem = i / DDIM;
    int h   = rem % HDIM;
    int e   = rem / HDIM;
    w2t[((size_t)e * DDIM + d) * HDIM + h] = f2bf(w2[i]);
}

// ---------------------------------------------------------------- router: one wave per token
__global__ __launch_bounds__(256)
void kRouter(const float* __restrict__ x, const float* __restrict__ wg,
             const float* __restrict__ biases,
             int* __restrict__ ridx, float* __restrict__ rgate) {
    int wave = threadIdx.x >> 5;
    int lane = threadIdx.x & 31;
    int t = blockIdx.x * 8 + wave;
    if (t >= M_TOK) return;
    float acc[8] = {};
    const float* xr = x + (size_t)t * DDIM;
    for (int d = lane; d < DDIM; d += 32) {
        float xv = xr[d];
        const float4* wr = (const float4*)(wg + (size_t)d * NEXP);
        float4 w0 = wr[0], w1 = wr[1];
        acc[0] += xv * w0.x; acc[1] += xv * w0.y; acc[2] += xv * w0.z; acc[3] += xv * w0.w;
        acc[4] += xv * w1.x; acc[5] += xv * w1.y; acc[6] += xv * w1.z; acc[7] += xv * w1.w;
    }
    #pragma unroll
    for (int n = 0; n < 8; n++) {
        #pragma unroll
        for (int off = 16; off > 0; off >>= 1)
            acc[n] += __shfl_xor(acc[n], off, 32);
    }
    if (lane == 0) {
        float best = -1e30f; int bi = 0;
        float s[8];
        #pragma unroll
        for (int n = 0; n < 8; n++) {
            s[n] = fastrcp(1.0f + __expf(-acc[n]));          // sigmoid via v_rcp_f32
            float v = s[n] + biases[n];                      // shared slot has -inf bias
            if (v > best) { best = v; bi = n; }
        }
        float w = s[bi];
        ridx[t]  = bi;
        rgate[t] = w * fastrcp(w + 1e-9f);                   // top-1 gate normalization
    }
}

// ---------------------------------------------------------------- build compacted expert lists
__global__ void kBuild(const int* __restrict__ ridx, const float* __restrict__ rgate,
                       int* __restrict__ cnt, int* __restrict__ list, float* __restrict__ gate) {
    int t = blockIdx.x * 256 + threadIdx.x;
    if (t >= M_TOK) return;
    int e = ridx[t];
    int p = atomicAdd(&cnt[e], 1);
    list[e * M_TOK + p] = t;
    gate[e * M_TOK + p] = rgate[t];
    int p2 = atomicAdd(&cnt[SHARED_E], 1);                   // shared expert: every token, w = 1
    list[SHARED_E * M_TOK + p2] = t;
    gate[SHARED_E * M_TOK + p2] = 1.0f;
}

// ---------------------------------------------------------------- GEMM1 + SwiGLU (bf16 WMMA)
// 128x64 block tile, 4 waves x (32 rows x 64 cols), double-buffered async LDS staging
__global__ __launch_bounds__(128)
void kGemm1(const unsigned short* __restrict__ xb, const unsigned short* __restrict__ w1t,
            const int* __restrict__ list, const int* __restrict__ cnt,
            unsigned short* __restrict__ hbuf) {
    const int e        = blockIdx.z;
    const int rowStart = blockIdx.y * 128;
    const int c0       = blockIdx.x * 64;                    // g-cols; u-cols are c0+1024
    const int count    = cnt[e];
    if (rowStart >= count) return;

    __shared__ unsigned short As[2][128 * LSTR];
    __shared__ unsigned short Bg[2][64 * LSTR];
    __shared__ unsigned short Bu[2][64 * LSTR];

    const int tid  = threadIdx.x;
    const int lane = tid & 31;
    const int wave = tid >> 5;
    const int lrow = lane & 15;
    const int kh   = lane >> 4;
    const int rb   = tid >> 1;                               // 0..63 weight row
    const int hb   = tid & 1;

    int rr = rowStart + tid;
    int tokA = (rr < count) ? list[e * M_TOK + rr] : list[e * M_TOK];
    const unsigned short* aRow = xb  + (size_t)tokA * DDIM;
    const unsigned short* gRow = w1t + ((size_t)e * TWOH + (c0 + rb)) * DDIM;
    const unsigned short* uRow = w1t + ((size_t)e * TWOH + (c0 + 1024 + rb)) * DDIM;

    auto stage = [&](int kk, int b) {
        #pragma unroll
        for (int c = 0; c < 32; c += 8)                      // A: 128x32 tile, 1 row/thread
            asyncB128(aRow + kk + c, &As[b][tid * LSTR + c]);
        asyncB128(gRow + kk + hb * 16,     &Bg[b][rb * LSTR + hb * 16]);
        asyncB128(gRow + kk + hb * 16 + 8, &Bg[b][rb * LSTR + hb * 16 + 8]);
        asyncB128(uRow + kk + hb * 16,     &Bu[b][rb * LSTR + hb * 16]);
        asyncB128(uRow + kk + hb * 16 + 8, &Bu[b][rb * LSTR + hb * 16 + 8]);
    };

    v8f accG[2][4] = {}, accU[2][4] = {};

    stage(0, 0);
    waitAsync0();
    __syncthreads();
    int cur = 0;

    for (int kk = 0; kk < DDIM; kk += 32) {
        const bool pre = (kk + 32) < DDIM;
        if (pre) stage(kk + 32, cur ^ 1);                    // prefetch overlaps WMMAs below

        Frag a0, a1;                                          // 2 A frags: rows wr..wr+15, wr+16..wr+31
        const int aB = (wave * 32 + lrow) * LSTR + kh * 8;
        a0.q[0] = *(const uint4*)&As[cur][aB];
        a0.q[1] = *(const uint4*)&As[cur][aB + 16];
        a1.q[0] = *(const uint4*)&As[cur][aB + 16 * LSTR];
        a1.q[1] = *(const uint4*)&As[cur][aB + 16 * LSTR + 16];

        #pragma unroll
        for (int j = 0; j < 4; j++) {                        // each B frag feeds 2 WMMAs
            Frag b;
            const int bB = (j * 16 + lrow) * LSTR + kh * 8;
            b.q[0] = *(const uint4*)&Bg[cur][bB];
            b.q[1] = *(const uint4*)&Bg[cur][bB + 16];
            accG[0][j] = WMMA_BF16(a0.v, b.v, accG[0][j]);
            accG[1][j] = WMMA_BF16(a1.v, b.v, accG[1][j]);
            b.q[0] = *(const uint4*)&Bu[cur][bB];
            b.q[1] = *(const uint4*)&Bu[cur][bB + 16];
            accU[0][j] = WMMA_BF16(a0.v, b.v, accU[0][j]);
            accU[1][j] = WMMA_BF16(a1.v, b.v, accU[1][j]);
        }

        if (pre) { waitAsync0(); __syncthreads(); }          // own prefetch landed + all reads done
        cur ^= 1;
    }

    // fused SwiGLU: h = silu(g) * u  (v_rcp_f32, not IEEE divide)
    #pragma unroll
    for (int m = 0; m < 2; m++) {
        #pragma unroll
        for (int j = 0; j < 4; j++) {
            #pragma unroll
            for (int i = 0; i < 8; i++) {
                int row = wave * 32 + m * 16 + i + ((lane >= 16) ? 8 : 0);
                int gr  = rowStart + row;
                if (gr < count) {
                    float g = accG[m][j][i], u = accU[m][j][i];
                    float h = g * fastrcp(1.0f + __expf(-g)) * u;
                    hbuf[((size_t)e * M_TOK + gr) * HDIM + (c0 + j * 16 + lrow)] = f2bf(h);
                }
            }
        }
    }
}

// ---------------------------------------------------------------- GEMM2: y += gate * (h @ W2)
__global__ __launch_bounds__(128)
void kGemm2(const unsigned short* __restrict__ hbuf, const unsigned short* __restrict__ w2t,
            const int* __restrict__ list, const float* __restrict__ gate,
            const int* __restrict__ cnt, float* __restrict__ y, int eBase) {
    const int e        = eBase + blockIdx.z;
    const int rowStart = blockIdx.y * 128;
    const int c0       = blockIdx.x * 64;
    const int count    = cnt[e];
    if (rowStart >= count) return;

    __shared__ unsigned short As[2][128 * LSTR];
    __shared__ unsigned short Bs[2][64 * LSTR];
    __shared__ int   toks[128];
    __shared__ float gs[128];

    const int tid  = threadIdx.x;
    const int lane = tid & 31;
    const int wave = tid >> 5;
    const int lrow = lane & 15;
    const int kh   = lane >> 4;
    const int rb   = tid >> 1;
    const int hb   = tid & 1;

    int rr = rowStart + tid;
    bool vld = rr < count;
    toks[tid] = vld ? list[e * M_TOK + rr] : 0;
    gs[tid]   = vld ? gate[e * M_TOK + rr] : 0.0f;

    int rc = vld ? rr : rowStart;
    const unsigned short* aRow = hbuf + ((size_t)e * M_TOK + rc) * HDIM;
    const unsigned short* bRow = w2t  + ((size_t)e * DDIM + (c0 + rb)) * HDIM;

    auto stage = [&](int kk, int b) {
        #pragma unroll
        for (int c = 0; c < 32; c += 8)
            asyncB128(aRow + kk + c, &As[b][tid * LSTR + c]);
        asyncB128(bRow + kk + hb * 16,     &Bs[b][rb * LSTR + hb * 16]);
        asyncB128(bRow + kk + hb * 16 + 8, &Bs[b][rb * LSTR + hb * 16 + 8]);
    };

    v8f acc[2][4] = {};

    stage(0, 0);
    waitAsync0();
    __syncthreads();
    int cur = 0;

    for (int kk = 0; kk < HDIM; kk += 32) {
        const bool pre = (kk + 32) < HDIM;
        if (pre) stage(kk + 32, cur ^ 1);

        Frag a0, a1;
        const int aB = (wave * 32 + lrow) * LSTR + kh * 8;
        a0.q[0] = *(const uint4*)&As[cur][aB];
        a0.q[1] = *(const uint4*)&As[cur][aB + 16];
        a1.q[0] = *(const uint4*)&As[cur][aB + 16 * LSTR];
        a1.q[1] = *(const uint4*)&As[cur][aB + 16 * LSTR + 16];

        #pragma unroll
        for (int j = 0; j < 4; j++) {
            Frag b;
            const int bB = (j * 16 + lrow) * LSTR + kh * 8;
            b.q[0] = *(const uint4*)&Bs[cur][bB];
            b.q[1] = *(const uint4*)&Bs[cur][bB + 16];
            acc[0][j] = WMMA_BF16(a0.v, b.v, acc[0][j]);
            acc[1][j] = WMMA_BF16(a1.v, b.v, acc[1][j]);
        }

        if (pre) { waitAsync0(); __syncthreads(); }
        cur ^= 1;
    }

    // each (token, d) touched exactly once per launch -> plain RMW, no atomics
    #pragma unroll
    for (int m = 0; m < 2; m++) {
        #pragma unroll
        for (int j = 0; j < 4; j++) {
            #pragma unroll
            for (int i = 0; i < 8; i++) {
                int row = wave * 32 + m * 16 + i + ((lane >= 16) ? 8 : 0);
                int gr  = rowStart + row;
                if (gr < count) {
                    int   t = toks[row];
                    float w = gs[row];
                    float* p = y + (size_t)t * DDIM + (c0 + j * 16 + lrow);
                    *p += w * acc[m][j][i];
                }
            }
        }
    }
}

// ---------------------------------------------------------------- host-side launch
extern "C" void kernel_launch(void* const* d_in, const int* in_sizes, int n_in,
                              void* d_out, int out_size, void* d_ws, size_t ws_size,
                              hipStream_t stream) {
    (void)in_sizes; (void)n_in; (void)out_size; (void)ws_size;
    const float* x      = (const float*)d_in[0];   // [M, D]
    const float* wg     = (const float*)d_in[1];   // [D, N]
    const float* w1     = (const float*)d_in[2];   // [N, D, 2H]
    const float* w2     = (const float*)d_in[3];   // [N, H, D]
    const float* biases = (const float*)d_in[4];   // [N]
    float* y = (float*)d_out;                      // [M, D] fp32

    char* ws = (char*)d_ws;
    size_t off = 0;
    auto take = [&](size_t bytes) { void* p = ws + off; off = (off + bytes + 255) & ~(size_t)255; return p; };
    unsigned short* xb   = (unsigned short*)take((size_t)M_TOK * DDIM * 2);
    unsigned short* w1t  = (unsigned short*)take((size_t)NEXP * TWOH * DDIM * 2);
    unsigned short* w2t  = (unsigned short*)take((size_t)NEXP * DDIM * HDIM * 2);
    unsigned short* hbuf = (unsigned short*)take((size_t)NEXP * M_TOK * HDIM * 2);
    int*   list  = (int*)  take((size_t)NEXP * M_TOK * 4);
    float* gate  = (float*)take((size_t)NEXP * M_TOK * 4);
    int*   ridx  = (int*)  take((size_t)M_TOK * 4);
    float* rgate = (float*)take((size_t)M_TOK * 4);
    int*   cnt   = (int*)  take((size_t)NEXP * 4);

    kInit<<<(M_TOK * DDIM / 4 + 255) / 256, 256, 0, stream>>>(y, cnt);
    kConvertX<<<(M_TOK * DDIM / 8 + 255) / 256, 256, 0, stream>>>(x, xb);
    kTransW1<<<(NEXP * DDIM * TWOH + 255) / 256, 256, 0, stream>>>(w1, w1t);
    kTransW2<<<(NEXP * HDIM * DDIM + 255) / 256, 256, 0, stream>>>(w2, w2t);
    kRouter<<<M_TOK / 8, 256, 0, stream>>>(x, wg, biases, ridx, rgate);
    kBuild<<<(M_TOK + 255) / 256, 256, 0, stream>>>(ridx, rgate, cnt, list, gate);
    kGemm1<<<dim3(HDIM / 64, M_TOK / 128, NEXP), 128, 0, stream>>>(xb, w1t, list, cnt, hbuf);
    kGemm2<<<dim3(DDIM / 64, M_TOK / 128, NEXP - 1), 128, 0, stream>>>(hbuf, w2t, list, gate, cnt, y, 0);
    kGemm2<<<dim3(DDIM / 64, M_TOK / 128, 1), 128, 0, stream>>>(hbuf, w2t, list, gate, cnt, y, SHARED_E);
}
